// S4DSSM_73143293051246
// MI455X (gfx1250) — matrix operand order repaired and checked by reading, not compiled, not run on hardware
//
#include <hip/hip_runtime.h>

// ---------------------------------------------------------------------------
// S4D SSM:  out = (decay ∘ cumsum(inv_decay ∘ (x·B^T))) · C^T
//   x [8, 8192, 512] f32,  B [128,512],  C [512,128],  scalar log_dt, lam[128]
// Strategy: WMMA f16 GEMM1 (+scale +tile cumsum) -> tiny cross-tile scan ->
//           WMMA f16 GEMM2 (+offset +decay). HBM-bound (~335 MB ≈ 14 µs at
//           23.3 TB/s); intermediate touched once each way. WS use ~34.1 MB.
// Round-3 change: __builtin_amdgcn_sched_barrier(0) between the operand-load
// block and the WMMA burst so all fragment loads issue as one clause and the
// waits become progressive instead of a full drain before every v_wmma.
// ---------------------------------------------------------------------------

#define TSEQ    8192
#define DMODEL  512
#define DSTATE  128
#define TILE_T  128
#define NTILE_T (TSEQ / TILE_T) /* 64 */

typedef __attribute__((ext_vector_type(16))) _Float16 v16h;
typedef __attribute__((ext_vector_type(8)))  _Float16 v8h;
typedef __attribute__((ext_vector_type(8)))  float    v8f;

// ---------------------------------------------------------------------------
// Kernel 0: repack B and C into per-lane WMMA B-operand fragment order (f16).
// Fragment layout assumption (16-bit B operand, 32x16 KxN, wave32):
//   lane&15 -> N, element e -> K = (e&7) + 8*(lane>>4) + 16*(e>>3).  Each lane's 16 halves are
//   stored contiguously (32B) so the hot loop loads them as 2x global b128.
// packB frag id = k0*8 + nt   (k0: 0..15 K-step, nt: 0..7 state tile)
// packC frag id = k0*32 + dt  (k0: 0..3  K-step, dt: 0..31 dmodel tile)
// ---------------------------------------------------------------------------
__global__ __launch_bounds__(256) void s4d_pack(const float* __restrict__ Bm,
                                                const float* __restrict__ Cm,
                                                _Float16* __restrict__ packB,
                                                _Float16* __restrict__ packC) {
  const int idx = blockIdx.x * 256 + threadIdx.x;
  if (idx < 16 * 8 * 32 * 16) {                       // 65536 B-frag elements
    const int e = idx & 15, lane = (idx >> 4) & 31, r = idx >> 9;
    const int nt = r & 7, k0 = r >> 3;
    packB[idx] = (_Float16)Bm[(nt * 16 + (lane & 15)) * DMODEL + (k0 * 32 + (e & 7) + 8 * (lane >> 4) + 16 * (e >> 3))];
  } else {                                            // 65536 C-frag elements
    const int i = idx - 65536;
    const int e = i & 15, lane = (i >> 4) & 31, r = i >> 9;
    const int dt = r & 31, k0 = r >> 5;
    // GEMM2 B-operand element [k=n][col=d] = C[d][n]
    packC[i] = (_Float16)Cm[(dt * 16 + (lane & 15)) * DSTATE + (k0 * 32 + (e & 7) + 8 * (lane >> 4) + 16 * (e >> 3))];
  }
}

// ---------------------------------------------------------------------------
// Kernel 1: Bu_scaled tile GEMM + intra-tile inclusive cumsum.
// Grid (NTILE_T, batch), block 256 = 8 waves. Wave w owns rows [16w,16w+16).
// ---------------------------------------------------------------------------
__global__ __launch_bounds__(256) void s4d_gemm1(
    const float* __restrict__ x, const float* __restrict__ log_lambda,
    const float* __restrict__ log_dt, const _Float16* __restrict__ packB,
    float* __restrict__ ws_bu, float* __restrict__ ws_sums) {
  extern __shared__ float smem[];                 // TILE_T*DSTATE f32 = 64 KB
  const int tid = threadIdx.x, lane = tid & 31, wave = tid >> 5;
  const int half = lane >> 4, mr = lane & 15;
  const int tile = blockIdx.x, b = blockIdx.y;
  const float dtv = __expf(log_dt[0]);

  const size_t rowbase = (size_t)b * TSEQ + (size_t)tile * TILE_T;
  const float* xrow = x + (rowbase + (size_t)(wave * 16 + mr)) * DMODEL;

  const v8f vzero = {0.f, 0.f, 0.f, 0.f, 0.f, 0.f, 0.f, 0.f};
  v8f acc[8];
#pragma unroll
  for (int nt = 0; nt < 8; ++nt) acc[nt] = vzero;

  for (int k0 = 0; k0 < DMODEL / 32; ++k0) {      // 16 K-steps of 32
    const int kb = k0 * 32;
    __builtin_prefetch(xrow + kb + 64, 0, 1);     // -> global_prefetch_b8

    // ---- Load block: 4x A (f32) + 16x B (f16 frags) issued together ------
    // A fragment: lane->M, elements 0..7 = K kb+8h.., 8..15 = K kb+16+8h..
    const float4 a0 = *(const float4*)(xrow + kb + half * 8);
    const float4 a1 = *(const float4*)(xrow + kb + half * 8 + 4);
    const float4 a2 = *(const float4*)(xrow + kb + 16 + half * 8);
    const float4 a3 = *(const float4*)(xrow + kb + 16 + half * 8 + 4);

    const v16h* bfr = (const v16h*)(packB + ((size_t)(k0 * 8) * 32 + lane) * 16);
    v16h Bf[8];
#pragma unroll
    for (int nt = 0; nt < 8; ++nt) Bf[nt] = bfr[nt * 32];  // 2x b128 each

    // Scheduling fence: keep all fragment loads above the compute burst so
    // the compiler emits one clause + progressive s_wait_loadcnt.
    __builtin_amdgcn_sched_barrier(0);

    // ---- Compute block: convert A (waits only on the 4 A loads), WMMAs ---
    v16h A;
    A[0]  = (_Float16)a0.x; A[1]  = (_Float16)a0.y;
    A[2]  = (_Float16)a0.z; A[3]  = (_Float16)a0.w;
    A[4]  = (_Float16)a1.x; A[5]  = (_Float16)a1.y;
    A[6]  = (_Float16)a1.z; A[7]  = (_Float16)a1.w;
    A[8]  = (_Float16)a2.x; A[9]  = (_Float16)a2.y;
    A[10] = (_Float16)a2.z; A[11] = (_Float16)a2.w;
    A[12] = (_Float16)a3.x; A[13] = (_Float16)a3.y;
    A[14] = (_Float16)a3.z; A[15] = (_Float16)a3.w;

#pragma unroll
    for (int nt = 0; nt < 8; ++nt) {
      acc[nt] = __builtin_amdgcn_wmma_f32_16x16x32_f16(
          false, A, false, Bf[nt], (short)0, acc[nt], false, false);
    }
  }

  // Scale by inv_decay and stage tile in LDS. C/D layout: VGPR j -> M=j (+8
  // for lane-half 1); lane&15 -> N.
#pragma unroll
  for (int nt = 0; nt < 8; ++nt) {
    const int n = nt * 16 + mr;
    const float ld = -__expf(log_lambda[n]) * dtv;   // log_decay[n] (<= 0)
#pragma unroll
    for (int j = 0; j < 8; ++j) {
      const int tl = wave * 16 + j + 8 * half;
      const float a = fmaxf((float)(tile * TILE_T + tl) * ld, -30.0f);
      const float inv = fminf(__expf(-a), 1.0e6f);
      smem[tl * DSTATE + n] = acc[nt][j] * inv;
    }
  }
  __syncthreads();

  // Intra-tile inclusive cumsum along t: one thread per state column.
  if (tid < DSTATE) {
    const int n = tid;
    float run = 0.0f;
    float* dst = ws_bu + rowbase * DSTATE + n;
    for (int t = 0; t < TILE_T; ++t) {
      run += smem[t * DSTATE + n];
      dst[(size_t)t * DSTATE] = run;
    }
    ws_sums[((size_t)b * NTILE_T + tile) * DSTATE + n] = run;  // tile total
  }
}

// ---------------------------------------------------------------------------
// Kernel 2: exclusive scan of tile totals across the 64 tiles (per batch, n).
// ---------------------------------------------------------------------------
__global__ __launch_bounds__(128) void s4d_scan(float* __restrict__ ws_sums) {
  const int b = blockIdx.x, n = threadIdx.x;
  float* p = ws_sums + (size_t)b * NTILE_T * DSTATE + n;
  float run = 0.0f;
  for (int t = 0; t < NTILE_T; ++t) {
    const float v = p[(size_t)t * DSTATE];
    p[(size_t)t * DSTATE] = run;                  // totals -> exclusive prefix
    run += v;
  }
}

// ---------------------------------------------------------------------------
// Kernel 3: h_out = (tile_cumsum + offset) * decay  ->  f16 LDS  ->  GEMM2.
// Grid (NTILE_T, batch), block 256 = 8 waves.
// ---------------------------------------------------------------------------
__global__ __launch_bounds__(256) void s4d_gemm2(
    const float* __restrict__ ws_bu, const float* __restrict__ ws_sums,
    const float* __restrict__ log_lambda, const float* __restrict__ log_dt,
    const _Float16* __restrict__ packC, float* __restrict__ out) {
  __shared__ _Float16 hsm[TILE_T * DSTATE];       // 32 KB
  __shared__ float ldec[DSTATE];
  const int tid = threadIdx.x, lane = tid & 31, wave = tid >> 5;
  const int half = lane >> 4, mr = lane & 15;
  const int tile = blockIdx.x, b = blockIdx.y;
  const float dtv = __expf(log_dt[0]);

  if (tid < DSTATE) ldec[tid] = -__expf(log_lambda[tid]) * dtv;
  __syncthreads();

  const size_t bubase = ((size_t)b * TSEQ + (size_t)tile * TILE_T) * DSTATE;
  const float* soff = ws_sums + ((size_t)b * NTILE_T + tile) * DSTATE;
#pragma unroll 4
  for (int i = 0; i < (TILE_T * DSTATE) / 256; ++i) {
    const int idx = i * 256 + tid;
    const int tl = idx >> 7, n = idx & (DSTATE - 1);
    const float hs = ws_bu[bubase + idx] + soff[n];
    const float a = fmaxf((float)(tile * TILE_T + tl) * ldec[n], -30.0f);
    hsm[idx] = (_Float16)(hs * __expf(a));
  }
  __syncthreads();

  const v8f vzero = {0.f, 0.f, 0.f, 0.f, 0.f, 0.f, 0.f, 0.f};
  float* orow = out + ((size_t)b * TSEQ + (size_t)tile * TILE_T +
                       (size_t)(wave * 16)) * DMODEL;
  const int tlA = wave * 16 + mr;                 // A-fragment row for lane

  for (int cg = 0; cg < 4; ++cg) {                // 4 column groups of 128 d
    v8f acc[8];
#pragma unroll
    for (int dtl = 0; dtl < 8; ++dtl) acc[dtl] = vzero;
#pragma unroll
    for (int k0 = 0; k0 < DSTATE / 32; ++k0) {    // 4 K-steps of 32
      const int kb = k0 * 32;
      // ---- Load block: 16x global b128 (C frags) + 2x ds_load_b128 (A) ---
      const v16h* cfr =
          (const v16h*)(packC + ((size_t)(k0 * 32 + cg * 8) * 32 + lane) * 16);
      v16h Cf[8];
#pragma unroll
      for (int dtl = 0; dtl < 8; ++dtl) Cf[dtl] = cfr[dtl * 32];

      const v8h a0 = *(const v8h*)&hsm[tlA * DSTATE + kb + 8 * half];
      const v8h a1 = *(const v8h*)&hsm[tlA * DSTATE + kb + 16 + 8 * half];

      __builtin_amdgcn_sched_barrier(0);

      // ---- Compute block ------------------------------------------------
      const v16h A = __builtin_shufflevector(a0, a1, 0, 1, 2, 3, 4, 5, 6, 7,
                                             8, 9, 10, 11, 12, 13, 14, 15);
#pragma unroll
      for (int dtl = 0; dtl < 8; ++dtl) {
        acc[dtl] = __builtin_amdgcn_wmma_f32_16x16x32_f16(
            false, A, false, Cf[dtl], (short)0, acc[dtl], false, false);
      }
    }
#pragma unroll
    for (int dtl = 0; dtl < 8; ++dtl) {
#pragma unroll
      for (int j = 0; j < 8; ++j) {
        const int tl = j + 8 * half;
        orow[(size_t)tl * DMODEL + cg * 128 + dtl * 16 + mr] = acc[dtl][j];
      }
    }
  }
}

// ---------------------------------------------------------------------------
// Host-side launch. Workspace layout (bytes):
//   ws_bu   : batch*8192*128*4      = 33,554,432
//   ws_sums : batch*64*128*4        =    262,144
//   packB   : 128*512*2             =    131,072
//   packC   : 512*128*2             =    131,072   (total ~34.1 MB)
// ---------------------------------------------------------------------------
extern "C" void kernel_launch(void* const* d_in, const int* in_sizes, int n_in,
                              void* d_out, int out_size, void* d_ws,
                              size_t ws_size, hipStream_t stream) {
  (void)n_in; (void)out_size; (void)ws_size;
  const float* x          = (const float*)d_in[0];
  const float* log_lambda = (const float*)d_in[1];
  const float* Bm         = (const float*)d_in[2];
  const float* Cm         = (const float*)d_in[3];
  const float* log_dt     = (const float*)d_in[4];
  float* out = (float*)d_out;

  const int batch = in_sizes[0] / (TSEQ * DMODEL);  // 8

  float* ws_bu   = (float*)d_ws;
  float* ws_sums = ws_bu + (size_t)batch * TSEQ * DSTATE;
  _Float16* packB = (_Float16*)(ws_sums + (size_t)batch * NTILE_T * DSTATE);
  _Float16* packC = packB + (size_t)DSTATE * DMODEL;

  s4d_pack<<<512, 256, 0, stream>>>(Bm, Cm, packB, packC);
  s4d_gemm1<<<dim3(NTILE_T, batch), 256, TILE_T * DSTATE * sizeof(float),
              stream>>>(x, log_lambda, log_dt, packB, ws_bu, ws_sums);
  s4d_scan<<<dim3(batch), 128, 0, stream>>>(ws_sums);
  s4d_gemm2<<<dim3(NTILE_T, batch), 256, 0, stream>>>(ws_bu, ws_sums,
                                                      log_lambda, log_dt,
                                                      packC, out);
}
